// MultiHeadAttention_59030030516306
// MI455X (gfx1250) — matrix operand order repaired and checked by reading, not compiled, not run on hardware
//
#include <hip/hip_runtime.h>
#include <hip/hip_bf16.h>

// ---------------- problem constants ----------------
#define BATCH 2
#define SEQ   2048
#define HID   2048
#define NH    16
#define HD    64
#define MTOT  (BATCH*SEQ)   // 4096

typedef __bf16 bf16;
typedef __attribute__((ext_vector_type(16))) __bf16 bf16x16;
typedef __attribute__((ext_vector_type(8)))  __bf16 bf16x8;
typedef __attribute__((ext_vector_type(8)))  float  f32x8;
typedef __attribute__((ext_vector_type(4)))  int    v4i;

// Async global->LDS path (CDNA5): gate on builtin availability so the file
// always compiles; fallback is the sync load+ds_store path.
#if defined(__has_builtin)
#  if __has_builtin(__builtin_amdgcn_global_load_async_to_lds_b128) && \
      __has_builtin(__builtin_amdgcn_s_wait_asynccnt)
#    define HAVE_ASYNC_LDS 1
#  endif
#endif
#ifndef HAVE_ASYNC_LDS
#  define HAVE_ASYNC_LDS 0
#endif

#if HAVE_ASYNC_LDS
typedef __attribute__((address_space(1))) v4i* gbl_v4i_ptr;
typedef __attribute__((address_space(3))) v4i* lds_v4i_ptr;
__device__ __forceinline__ void async_cp_b128(const bf16* g, bf16* l) {
  __builtin_amdgcn_global_load_async_to_lds_b128(
      (gbl_v4i_ptr)const_cast<bf16*>(g),
      (lds_v4i_ptr)l,
      /*offset=*/0, /*cpol=*/0);
}
#endif

// A-fragment (16x32 bf16, M per lane): lane holds row M=lane%16,
// k = h*16 .. h*16+15 contiguous (h=lane/16).
// Caller passes p = rowBase + h*16 (16B aligned).
__device__ __forceinline__ bf16x16 frag_a(const bf16* p) {
  bf16x8 lo = *(const bf16x8*)p;
  bf16x8 hi = *(const bf16x8*)(p + 8);
  return __builtin_shufflevector(lo, hi, 0,1,2,3,4,5,6,7,8,9,10,11,12,13,14,15);
}

// B-fragment (32x16 bf16, N per lane): lane holds column N=lane%16,
// k = laneHalf*16 .. +15 contiguous. Caller passes p = colBase + h*16.
__device__ __forceinline__ bf16x16 frag_b(const bf16* p) {
  bf16x8 lo = *(const bf16x8*)p;
  bf16x8 hi = *(const bf16x8*)(p + 8);
  return __builtin_shufflevector(lo, hi, 0,1,2,3,4,5,6,7,8,9,10,11,12,13,14,15);
}

__device__ __forceinline__ f32x8 wmma_bf16(bf16x16 a, bf16x16 b, f32x8 c) {
  return __builtin_amdgcn_wmma_f32_16x16x32_bf16(false, a, false, b, (short)0, c, false, false);
}

// ---------------- conversion kernels ----------------
__global__ __launch_bounds__(256) void to_bf16_kernel(const float* __restrict__ in,
                                                      bf16* __restrict__ out, int n) {
  int i = blockIdx.x * 256 + threadIdx.x;
  if (i < n) out[i] = (bf16)in[i];
}

// W[k][n] f32 (row-major [K,N]=[2048,2048]) -> Wt[n][k] bf16
__global__ __launch_bounds__(256) void transpose_to_bf16_kernel(const float* __restrict__ W,
                                                                bf16* __restrict__ Wt) {
  __shared__ bf16 tile[64][65];
  int bx = blockIdx.x * 64;  // n
  int by = blockIdx.y * 64;  // k
  for (int i = 0; i < 16; ++i) {
    int e = threadIdx.x + i * 256;
    int r = e >> 6, c = e & 63;            // r: k offset, c: n offset
    tile[c][r] = (bf16)W[(size_t)(by + r) * HID + bx + c];
  }
  __syncthreads();
  for (int i = 0; i < 16; ++i) {
    int e = threadIdx.x + i * 256;
    int r = e >> 6, c = e & 63;            // r: n offset, c: k offset
    Wt[(size_t)(bx + r) * HID + by + c] = tile[r][c];
  }
}

// ---------------- tiled bf16 GEMM, f32 accumulate ----------------
// C[M,N] = A[M,K] * Bt[N,K]^T ; M=4096, N=K=2048
// MODE 0: write f32 row-major to Cf
// MODE 1: write bf16 scattered into [b][h][s][d] layout (QKV)
// MODE 2: like MODE 1 but scaled by 1/sqrt(HD)=0.125 (exact in bf16; for Q)
#define BM 128
#define BN 128
#define BK 32
#define LDA 40   // LDS row stride (bf16 elems), 80B = 16B-aligned, conflict padding
#define KSTEPS (HID / BK)

template <int MODE>
__global__ __launch_bounds__(256)
void gemm_bf16_kernel(const bf16* __restrict__ A, const bf16* __restrict__ Bt,
                      float* __restrict__ Cf, bf16* __restrict__ Cq) {
#if HAVE_ASYNC_LDS
  __shared__ bf16 sA[2][BM * LDA];
  __shared__ bf16 sB[2][BN * LDA];
#else
  __shared__ bf16 sA[1][BM * LDA];
  __shared__ bf16 sB[1][BN * LDA];
#endif

  const int K = HID, N = HID;
  const int tid   = threadIdx.x;
  const int wave  = tid >> 5, lane = tid & 31;
  const int laneM = lane & 15, laneH = lane >> 4;
  const int waveM = wave & 1;        // 2 waves over M (64 rows each)
  const int waveN = wave >> 1;       // 4 waves over N (32 cols each)
  const int blockM = blockIdx.x * BM;
  const int blockN = blockIdx.y * BN;

  // staging coordinates: thread covers chunks c = tid*2, tid*2+1 of each tile
  const int c0 = tid * 2;
  const int row0 = c0 >> 2,       kc0 = (c0 & 3) * 8;
  const int row1 = (c0 + 1) >> 2, kc1 = ((c0 + 1) & 3) * 8;

  f32x8 acc[4][2] = {};

  auto compute = [&](const bf16* tA, const bf16* tB) {
    bf16x16 af[4], bfr[2];
#pragma unroll
    for (int mi = 0; mi < 4; ++mi)
      af[mi] = frag_a(tA + (waveM * 64 + mi * 16 + laneM) * LDA + laneH * 16);
#pragma unroll
    for (int ni = 0; ni < 2; ++ni)
      bfr[ni] = frag_b(tB + (waveN * 32 + ni * 16 + laneM) * LDA + laneH * 16);
#pragma unroll
    for (int mi = 0; mi < 4; ++mi)
#pragma unroll
      for (int ni = 0; ni < 2; ++ni)
        acc[mi][ni] = wmma_bf16(af[mi], bfr[ni], acc[mi][ni]);
  };

#if HAVE_ASYNC_LDS
  auto issue = [&](int k0, int buf) {
    async_cp_b128(A  + (size_t)(blockM + row0) * K + k0 + kc0, &sA[buf][row0 * LDA + kc0]);
    async_cp_b128(A  + (size_t)(blockM + row1) * K + k0 + kc1, &sA[buf][row1 * LDA + kc1]);
    async_cp_b128(Bt + (size_t)(blockN + row0) * K + k0 + kc0, &sB[buf][row0 * LDA + kc0]);
    async_cp_b128(Bt + (size_t)(blockN + row1) * K + k0 + kc1, &sB[buf][row1 * LDA + kc1]);
  };
  issue(0, 0);
  for (int it = 0; it < KSTEPS; ++it) {
    const int buf = it & 1;
    if (it + 1 < KSTEPS) {
      issue((it + 1) * BK, buf ^ 1);
      __builtin_amdgcn_s_wait_asynccnt(4);   // stage `it` complete (in-order)
    } else {
      __builtin_amdgcn_s_wait_asynccnt(0);
    }
    __syncthreads();                          // tile visible to all waves
    compute(sA[buf], sB[buf]);
    __syncthreads();                          // all reads done before refill
  }
#else
  for (int it = 0; it < KSTEPS; ++it) {
    const int k0 = it * BK;
    *(bf16x8*)(&sA[0][row0 * LDA + kc0]) = *(const bf16x8*)(A  + (size_t)(blockM + row0) * K + k0 + kc0);
    *(bf16x8*)(&sA[0][row1 * LDA + kc1]) = *(const bf16x8*)(A  + (size_t)(blockM + row1) * K + k0 + kc1);
    *(bf16x8*)(&sB[0][row0 * LDA + kc0]) = *(const bf16x8*)(Bt + (size_t)(blockN + row0) * K + k0 + kc0);
    *(bf16x8*)(&sB[0][row1 * LDA + kc1]) = *(const bf16x8*)(Bt + (size_t)(blockN + row1) * K + k0 + kc1);
    __syncthreads();
    compute(sA[0], sB[0]);
    __syncthreads();
  }
#endif

  // writeback: D element r -> row = 8*laneH + r, col = laneM (per ISA C/D layout)
#pragma unroll
  for (int mi = 0; mi < 4; ++mi)
#pragma unroll
    for (int ni = 0; ni < 2; ++ni) {
      int col = blockN + waveN * 32 + ni * 16 + laneM;
#pragma unroll
      for (int r = 0; r < 8; ++r) {
        int row = blockM + waveM * 64 + mi * 16 + laneH * 8 + r;
        float v = acc[mi][ni][r];
        if (MODE == 0) {
          Cf[(size_t)row * N + col] = v;
        } else {
          if (MODE == 2) v *= 0.125f;   // fold attention scale into Q (exact pow2)
          int b = row >> 11, s = row & (SEQ - 1);
          int h = col >> 6,  d = col & (HD - 1);
          Cq[(((size_t)(b * NH + h) * SEQ) + s) * HD + d] = (bf16)v;
        }
      }
    }
}

// ---------------- flash attention (causal), one wave per 16-query tile ----------------
// Q,K,V: bf16 [B*NH][SEQ][HD]; ctx: bf16 [B*S][HID] row-major.
// Q is pre-scaled by 1/sqrt(HD), so scores need no per-tile multiply.
__global__ __launch_bounds__(128)
void flash_attn_kernel(const bf16* __restrict__ Q, const bf16* __restrict__ Kb,
                       const bf16* __restrict__ V, bf16* __restrict__ ctx) {
  __shared__ bf16 sP[4][16 * 32];        // P tile per wave
  __shared__ bf16 sV[4][HD * LDA];       // transposed V tile per wave: Vt[d][key]

  const int wave = threadIdx.x >> 5, lane = threadIdx.x & 31;
  const int laneM = lane & 15, laneH = lane >> 4;
  const int bh = blockIdx.y;
  const int qBase = blockIdx.x * 64 + wave * 16;

  const bf16* Qh = Q  + (size_t)bh * SEQ * HD;
  const bf16* Kh = Kb + (size_t)bh * SEQ * HD;
  const bf16* Vh = V  + (size_t)bh * SEQ * HD;

  // Q fragments (A-type, 16 queries x 32 dims, two d-halves)
  const bf16* qrow = Qh + (size_t)(qBase + laneM) * HD + laneH * 16;
  bf16x16 aq0 = frag_a(qrow);
  bf16x16 aq1 = frag_a(qrow + 32);

  f32x8 o[4] = {};                       // 4 d-tiles of 16
  float m_[8], l_[8], ls[8];
#pragma unroll
  for (int r = 0; r < 8; ++r) { m_[r] = -1e30f; l_[r] = 0.f; }

  bf16* P  = sP[wave];
  bf16* Vt = sV[wave];
  const int kTiles = qBase / 32 + 1;

  for (int kt = 0; kt < kTiles; ++kt) {
    const int kb = kt * 32;

    if (kt + 1 < kTiles) {               // prefetch next tile's K/V rows
      __builtin_prefetch(Kh + (size_t)(kb + 32 + lane) * HD, 0, 1);
      __builtin_prefetch(Vh + (size_t)(kb + 32 + lane) * HD, 0, 1);
    }

    // stage V tile transposed: lane owns key row kb+lane, scatter over d
    {
      const bf16* vrow = Vh + (size_t)(kb + lane) * HD;
#pragma unroll
      for (int c = 0; c < 8; ++c) {
        bf16x8 v = *(const bf16x8*)(vrow + c * 8);
#pragma unroll
        for (int e = 0; e < 8; ++e) Vt[(c * 8 + e) * LDA + lane] = v[e];
      }
    }

    // scores: two 16-key column tiles, contraction over d (2 WMMAs each)
    f32x8 sc[2];
#pragma unroll
    for (int c = 0; c < 2; ++c) {
      const bf16* krow = Kh + (size_t)(kb + c * 16 + laneM) * HD;
      bf16x16 bk0 = frag_b(krow + laneH * 16);        // d 0..31
      bf16x16 bk1 = frag_b(krow + 32 + laneH * 16);   // d 32..63
      f32x8 s = {};
      s = wmma_bf16(aq0, bk0, s);
      s = wmma_bf16(aq1, bk1, s);
      sc[c] = s;
    }

    // online softmax: row = qBase + 8*laneH + r, cols laneM and laneM+16.
    // Only the final tile can cross the causal diagonal; full tiles skip masking.
    auto softmax_update = [&](bool domask) {
#pragma unroll
      for (int r = 0; r < 8; ++r) {
        float s0 = sc[0][r];
        float s1 = sc[1][r];
        if (domask) {
          int q = qBase + laneH * 8 + r;
          if (kb + laneM      > q) s0 = -1e30f;
          if (kb + 16 + laneM > q) s1 = -1e30f;
        }
        float tmax = fmaxf(s0, s1);
        for (int off = 1; off < 16; off <<= 1)
          tmax = fmaxf(tmax, __shfl_xor(tmax, off, 32));
        float mn = fmaxf(m_[r], tmax);
        float p0 = __expf(s0 - mn);
        float p1 = __expf(s1 - mn);
        float rs = p0 + p1;
        for (int off = 1; off < 16; off <<= 1)
          rs += __shfl_xor(rs, off, 32);
        float co = __expf(m_[r] - mn);
        l_[r] = l_[r] * co + rs;
        m_[r] = mn;
        ls[r] = co;
        P[(laneH * 8 + r) * 32 + laneM]      = (bf16)p0;
        P[(laneH * 8 + r) * 32 + 16 + laneM] = (bf16)p1;
      }
    };
    if (kt + 1 == kTiles) softmax_update(true);   // partial (diagonal) tile
    else                  softmax_update(false);  // fully unmasked tile

#pragma unroll
    for (int dt = 0; dt < 4; ++dt)
#pragma unroll
      for (int r = 0; r < 8; ++r) o[dt][r] *= ls[r];

    // ensure this wave's P/Vt stores are complete before fragment reload
    asm volatile("s_wait_dscnt 0" ::: "memory");

    bf16x16 pf = frag_a(P + laneM * 32 + laneH * 16);  // A-type: 16q x 32k
#pragma unroll
    for (int dt = 0; dt < 4; ++dt) {
      bf16x16 vf = frag_b(Vt + (dt * 16 + laneM) * LDA + laneH * 16);
      o[dt] = wmma_bf16(pf, vf, o[dt]);
    }
  }

  // epilogue: normalize and write ctx [b*S+s][h*64+d]
  const int b = bh >> 4, h = bh & (NH - 1);
#pragma unroll
  for (int dt = 0; dt < 4; ++dt)
#pragma unroll
    for (int r = 0; r < 8; ++r) {
      int s = qBase + laneH * 8 + r;
      float v = o[dt][r] / l_[r];
      ctx[((size_t)(b * SEQ + s)) * HID + h * HD + dt * 16 + laneM] = (bf16)v;
    }
}

// ---------------- host-side launch ----------------
extern "C" void kernel_launch(void* const* d_in, const int* in_sizes, int n_in,
                              void* d_out, int out_size, void* d_ws, size_t ws_size,
                              hipStream_t stream) {
  const float* x  = (const float*)d_in[0];
  const float* Wq = (const float*)d_in[1];
  const float* Wk = (const float*)d_in[2];
  const float* Wv = (const float*)d_in[3];
  const float* Wo = (const float*)d_in[4];
  float* out = (float*)d_out;

  // workspace carve-up (~112 MiB)
  char* ws = (char*)d_ws;
  size_t off = 0;
  bf16* xb  = (bf16*)(ws + off); off += (size_t)MTOT * HID * 2;
  bf16* WqT = (bf16*)(ws + off); off += (size_t)HID * HID * 2;
  bf16* WkT = (bf16*)(ws + off); off += (size_t)HID * HID * 2;
  bf16* WvT = (bf16*)(ws + off); off += (size_t)HID * HID * 2;
  bf16* WoT = (bf16*)(ws + off); off += (size_t)HID * HID * 2;
  bf16* Qb  = (bf16*)(ws + off); off += (size_t)MTOT * HID * 2;
  bf16* Kb  = (bf16*)(ws + off); off += (size_t)MTOT * HID * 2;
  bf16* Vb  = (bf16*)(ws + off); off += (size_t)MTOT * HID * 2;
  bf16* ctx = (bf16*)(ws + off); off += (size_t)MTOT * HID * 2;

  // 1) precision conversion
  to_bf16_kernel<<<(MTOT * HID) / 256, 256, 0, stream>>>(x, xb, MTOT * HID);
  dim3 tg(HID / 64, HID / 64);
  transpose_to_bf16_kernel<<<tg, 256, 0, stream>>>(Wq, WqT);
  transpose_to_bf16_kernel<<<tg, 256, 0, stream>>>(Wk, WkT);
  transpose_to_bf16_kernel<<<tg, 256, 0, stream>>>(Wv, WvT);
  transpose_to_bf16_kernel<<<tg, 256, 0, stream>>>(Wo, WoT);

  // 2) Q/K/V projections (bf16 WMMA GEMM, scatter into [b,h,s,d]); Q pre-scaled
  dim3 gg(MTOT / BM, HID / BN);
  gemm_bf16_kernel<2><<<gg, 256, 0, stream>>>(xb, WqT, nullptr, Qb);
  gemm_bf16_kernel<1><<<gg, 256, 0, stream>>>(xb, WkT, nullptr, Kb);
  gemm_bf16_kernel<1><<<gg, 256, 0, stream>>>(xb, WvT, nullptr, Vb);

  // 3) causal flash attention
  flash_attn_kernel<<<dim3(SEQ / 64, BATCH * NH), 128, 0, stream>>>(Qb, Kb, Vb, ctx);

  // 4) output projection, fp32 result
  gemm_bf16_kernel<0><<<gg, 256, 0, stream>>>(ctx, WoT, out, nullptr);
}
